// AttentionBlock2D_22557168239022
// MI455X (gfx1250) — matrix operand order repaired and checked by reading, not compiled, not run on hardware
//
#include <hip/hip_runtime.h>

// MI455X (gfx1250) fused attention block, round 6.
// ~19.3 GFLOP vs ~16 MB HBM -> memory/latency bound. Flash-attention core in
// bf16 WMMA 16x16x32 (f32 accum), softmax in f32 (exp2 domain, Q pre-scaled
// by log2(e)). Wave32; each wave owns 16 query columns; 64 keys per iteration.
//
// Layout tricks:
//  * Key-row permutation: within each 32-key group, St tile0 A-rows hold keys
//    {+0..7, +16..23}, tile1 {+8..15, +24..31}; exp(St) per lane is then
//    ALREADY the B-operand layout of P^T -> no cross-lane traffic for PV.
//  * K A-operand garbage in half-1 lanes / upper K maps to hardware-K >= 8,
//    which multiplies Q's zero rows -> all lanes load real addresses.
//  * softmax denominator l accumulated by a constant ones-row WMMA tile
//    (same bf16 P the numerator uses) instead of a 32-add VALU tree.
//  * K loads pipelined one chunk ahead; V loads hoisted above the softmax.

typedef __attribute__((ext_vector_type(16))) __bf16 v16bf;
typedef __attribute__((ext_vector_type(8)))  __bf16 v8bf;
typedef __attribute__((ext_vector_type(8)))  float  v8f;
typedef int v4i_t __attribute__((vector_size(16)));   // async-LDS builtin pointee

namespace {
constexpr int kB = 8;     // batch
constexpr int kC = 64;    // channels
constexpr int kN = 4096;  // H*W
constexpr int kD = 8;     // Dqk
constexpr float kLog2e = 1.44269504088896340736f;
}

#if __has_builtin(__builtin_amdgcn_global_load_async_to_lds_b128) && \
    __has_builtin(__builtin_amdgcn_s_wait_asynccnt)
#define HAVE_ASYNC_LDS 1
#endif
#if __has_builtin(__builtin_amdgcn_permlanex16)
#define HAVE_PERMX16 1
#endif
#if __has_builtin(__builtin_amdgcn_exp2f)
#define EXP2F(x) __builtin_amdgcn_exp2f(x)
#else
#define EXP2F(x) exp2f(x)
#endif

static __device__ __forceinline__ unsigned short f2bf(float f) {
  union { float f; unsigned u; } v; v.f = f;
  unsigned r = v.u + 0x7FFFu + ((v.u >> 16) & 1u);  // round-to-nearest-even
  return (unsigned short)(r >> 16);
}

// Pack two f32 into a bf16x2 dword. Native packed convert if present, else
// round-nearest-ties-up (inputs are probabilities in [0,1]; never NaN/Inf):
// 2 adds + shift + and_or (fusable to v_perm_b32) vs ~12 ops for exact RNE.
#if __has_builtin(__builtin_amdgcn_cvt_pk_bf16_f32)
static __device__ __forceinline__ unsigned pack2bf(float a, float b) {
  return __builtin_bit_cast(unsigned, __builtin_amdgcn_cvt_pk_bf16_f32(a, b));
}
#else
static __device__ __forceinline__ unsigned pack2bf(float a, float b) {
  unsigned au = __builtin_bit_cast(unsigned, a) + 0x8000u;
  unsigned bu = __builtin_bit_cast(unsigned, b) + 0x8000u;
  return (au >> 16) | (bu & 0xffff0000u);
}
#endif

// Exchange value with lane^16 (half-wave swap) without LDS if possible.
static __device__ __forceinline__ float xhalf_f32(float v) {
#ifdef HAVE_PERMX16
  unsigned u = __builtin_bit_cast(unsigned, v);
  unsigned o = __builtin_amdgcn_permlanex16(u, u, 0x76543210u, 0xfedcba98u,
                                            false, false);
  return __builtin_bit_cast(float, o);
#else
  return __shfl_xor(v, 16, 32);
#endif
}

union BfOp {
  v16bf v;
  v8bf h[2];
  unsigned u[8];
};

// ---------------------------------------------------------------------------
// Phase 1: 1x1-conv projections. q (pre-scaled by log2e), k: [B][N][8] bf16;
// v transposed: [B][C][N] bf16. x tile staged via async global->LDS copies
// (ASYNCcnt path); weight reads wave-uniform -> scalar loads.
// ---------------------------------------------------------------------------
__global__ __launch_bounds__(256) void proj_qkv_kernel(
    const float* __restrict__ x,
    const float* __restrict__ Wq, const float* __restrict__ bq,
    const float* __restrict__ Wk, const float* __restrict__ bk,
    const float* __restrict__ Wv, const float* __restrict__ bv,
    unsigned short* __restrict__ qws, unsigned short* __restrict__ kws,
    unsigned short* __restrict__ vtws, unsigned short* __restrict__ zws) {
  __shared__ float xs[kC][64];
  const int b  = blockIdx.y;
  const int n0 = blockIdx.x * 64;
  const int t  = threadIdx.x;
  const float* xb = x + (size_t)b * kC * kN;

  // One block writes the 16-byte zero pad used by the attention kernel (Q op).
  if (blockIdx.x == 0 && blockIdx.y == 0 && t < 8) zws[t] = 0;

#ifdef HAVE_ASYNC_LDS
#pragma unroll
  for (int i = 0; i < 4; ++i) {           // 1024 x 16B, coalesced
    int idx = t + i * 256;
    int c = idx >> 4, f4 = idx & 15;
    __builtin_amdgcn_global_load_async_to_lds_b128(
        (__attribute__((address_space(1))) v4i_t*)(xb + (size_t)c * kN + n0 + f4 * 4),
        (__attribute__((address_space(3))) v4i_t*)(&xs[c][f4 * 4]), 0, 0);
  }
  __builtin_amdgcn_s_wait_asynccnt(0);
#else
#pragma unroll
  for (int i = 0; i < 16; ++i) {
    int idx = t + i * 256;
    int c = idx >> 6, n = idx & 63;
    xs[c][n] = xb[(size_t)c * kN + n0 + n];
  }
#endif
  __syncthreads();

  const int n = t & 63;
  const int g = t >> 6;                   // wave-uniform group 0..3

  for (int ii = 0; ii < 16; ++ii) {       // V: 16 channels per group
    int co = g * 16 + ii;
    float acc = bv[co];
    const float* wr = Wv + co * kC;
#pragma unroll 8
    for (int c = 0; c < kC; ++c) acc = fmaf(wr[c], xs[c][n], acc);
    vtws[((size_t)b * kC + co) * kN + n0 + n] = f2bf(acc);
  }
  if (g == 0) {
    for (int d = 0; d < kD; ++d) {
      float acc = bq[d];
      const float* wr = Wq + d * kC;
#pragma unroll 8
      for (int c = 0; c < kC; ++c) acc = fmaf(wr[c], xs[c][n], acc);
      qws[((size_t)b * kN + n0 + n) * kD + d] = f2bf(acc * kLog2e);
    }
  } else if (g == 1) {
    for (int d = 0; d < kD; ++d) {
      float acc = bk[d];
      const float* wr = Wk + d * kC;
#pragma unroll 8
      for (int c = 0; c < kC; ++c) acc = fmaf(wr[c], xs[c][n], acc);
      kws[((size_t)b * kN + n0 + n) * kD + d] = f2bf(acc);
    }
  }
}

// ---------------------------------------------------------------------------
// Phase 2: flash attention + epilogue. 128 threads = 4 waves; 16 queries per
// wave; grid (N/64, B). Per 64-key chunk: 4 WMMA (St) + online softmax (exp2)
// + 10 WMMA (O^T += V^T P^T, plus ones-row tile accumulating l). O rescale
// skipped by a wave-uniform branch when no lane's running max moved.
// ---------------------------------------------------------------------------
__global__ __launch_bounds__(128) void flash_attn_kernel(
    const float* __restrict__ x, const float* __restrict__ gamma,
    const unsigned short* __restrict__ qws, const unsigned short* __restrict__ kws,
    const unsigned short* __restrict__ vtws, const unsigned short* __restrict__ zws,
    float* __restrict__ out) {
  const int b    = blockIdx.y;
  const int lane = threadIdx.x & 31;
  const int wv   = threadIdx.x >> 5;
  const int half = lane >> 4;
  const int r    = lane & 15;
  const int q    = blockIdx.x * 64 + wv * 16 + r;

  const unsigned short* kbp = kws  + (size_t)b * kN * kD;
  const unsigned short* vbp = vtws + (size_t)b * kC * kN;

  // Permuted key-row offset: A-rows 0..7 -> keys +0..7, rows 8..15 -> +16..23.
  // All lanes load real K data; half-1 lanes' elements hit Q's zero rows.
  const int koff = r + (r & 8);
  const unsigned short* kq = kbp + (size_t)koff * kD;

  // V row base for this lane; ct tiles and key groups are immediate offsets.
  const unsigned short* vq = vbp + (size_t)r * kN + 16 * half;

  // B-operand: Q^T (Dqk=8 zero-padded to 32); half-1 lanes MUST read zeros
  // (these zeros annihilate the K-operand garbage at hardware-K >= 8).
  const unsigned short* qp = half ? zws : qws + ((size_t)b * kN + q) * kD;
  BfOp qop;
#pragma unroll
  for (int i = 0; i < 16; ++i) qop.v[i] = (__bf16)0.0f;
  qop.h[0] = *(const v8bf*)qp;

  BfOp ka0, ka1, ka2, ka3;                // K A-operands; h[1] stays zero
#pragma unroll
  for (int i = 0; i < 16; ++i) {
    ka0.v[i] = (__bf16)0.0f; ka1.v[i] = (__bf16)0.0f;
    ka2.v[i] = (__bf16)0.0f; ka3.v[i] = (__bf16)0.0f;
  }

  // Constant ones-row A-operand: row 0 (lane r==0, both halves) = 1.0,
  // all other rows zero. Accumulates l = sum_k P into Ot4[0] (half-0 lanes).
  BfOp ones;
  {
    unsigned oval = (r == 0) ? 0x3F803F80u : 0u;
#pragma unroll
    for (int i = 0; i < 8; ++i) ones.u[i] = oval;
  }

  v8f zc;
#pragma unroll
  for (int j = 0; j < 8; ++j) zc[j] = 0.0f;

  v8f Ot[4], Ot4;
#pragma unroll
  for (int t = 0; t < 4; ++t)
#pragma unroll
    for (int j = 0; j < 8; ++j) Ot[t][j] = 0.0f;
#pragma unroll
  for (int j = 0; j < 8; ++j) Ot4[j] = 0.0f;

  float m_run = -3.0e38f;

  // Preload first K chunk.
  ka0.h[0] = *(const v8bf*)(kq);
  ka1.h[0] = *(const v8bf*)(kq + 8 * kD);
  ka2.h[0] = *(const v8bf*)(kq + 32 * kD);
  ka3.h[0] = *(const v8bf*)(kq + 40 * kD);

  for (int kb = 0; kb < kN; kb += 64) {
    // St logits (log2 domain); lane(half,r) vgpr j = key kb + g*32 + off.
    v8f s0 = __builtin_amdgcn_wmma_f32_16x16x32_bf16(false, ka0.v, false, qop.v,
                                                     (short)0, zc, false, false);
    v8f s1 = __builtin_amdgcn_wmma_f32_16x16x32_bf16(false, ka1.v, false, qop.v,
                                                     (short)0, zc, false, false);
    v8f s2 = __builtin_amdgcn_wmma_f32_16x16x32_bf16(false, ka2.v, false, qop.v,
                                                     (short)0, zc, false, false);
    v8f s3 = __builtin_amdgcn_wmma_f32_16x16x32_bf16(false, ka3.v, false, qop.v,
                                                     (short)0, zc, false, false);

    // Pipeline: next K chunk (tail over-read stays inside the workspace).
    {
      const unsigned short* kr = kq + (size_t)(kb + 64) * kD;
      ka0.h[0] = *(const v8bf*)(kr);
      ka1.h[0] = *(const v8bf*)(kr + 8 * kD);
      ka2.h[0] = *(const v8bf*)(kr + 32 * kD);
      ka3.h[0] = *(const v8bf*)(kr + 40 * kD);
    }

    // V tiles for this chunk: issue before the softmax VALU work.
    BfOp va[4][2];
    {
      const unsigned short* vrow = vq + kb;
#pragma unroll
      for (int ct = 0; ct < 4; ++ct) {
        const unsigned short* p = vrow + (size_t)ct * 16 * kN;
        va[ct][0].h[0] = *(const v8bf*)(p);        // k = 16*half + 0..7
        va[ct][0].h[1] = *(const v8bf*)(p + 8);    // k = 16*half + 8..15
        va[ct][1].h[0] = *(const v8bf*)(p + 32);
        va[ct][1].h[1] = *(const v8bf*)(p + 40);
      }
    }
    __builtin_prefetch(vq + kb + 64, 0, 1);        // global_prefetch_b8

    // Running max (two independent chains for ILP).
    float ma = fmaxf(s0[0], s1[0]);
    float mb = fmaxf(s2[0], s3[0]);
#pragma unroll
    for (int j = 1; j < 8; ++j) {
      ma = fmaxf(ma, fmaxf(s0[j], s1[j]));
      mb = fmaxf(mb, fmaxf(s2[j], s3[j]));
    }
    float mloc = fmaxf(ma, mb);
    mloc = fmaxf(mloc, xhalf_f32(mloc));
    float m_new = fmaxf(m_run, mloc);

    float p0[8], p1[8], p2[8], p3[8];
#pragma unroll
    for (int j = 0; j < 8; ++j) {
      p0[j] = EXP2F(s0[j] - m_new);
      p1[j] = EXP2F(s1[j] - m_new);
      p2[j] = EXP2F(s2[j] - m_new);
      p3[j] = EXP2F(s3[j] - m_new);
    }

    if (__any(m_new > m_run)) {           // wave-uniform: rescale only if needed
      float alpha = EXP2F(m_run - m_new);
#pragma unroll
      for (int t = 0; t < 4; ++t)
#pragma unroll
        for (int j = 0; j < 8; ++j) Ot[t][j] *= alpha;
      Ot4[0] *= alpha;                    // l rides in Ot4[0]
    }
    m_run = m_new;

    // P^T B-operands (lane-local thanks to key-row permutation).
    BfOp pop0, pop1;
#pragma unroll
    for (int i = 0; i < 4; ++i) {
      pop0.u[i]     = pack2bf(p0[2 * i], p0[2 * i + 1]);
      pop0.u[4 + i] = pack2bf(p1[2 * i], p1[2 * i + 1]);
      pop1.u[i]     = pack2bf(p2[2 * i], p2[2 * i + 1]);
      pop1.u[4 + i] = pack2bf(p3[2 * i], p3[2 * i + 1]);
    }

    // O^T[c][q] += V^T(c x m) * P^T(m x q); ones tile accumulates l.
#pragma unroll
    for (int ct = 0; ct < 4; ++ct) {
      Ot[ct] = __builtin_amdgcn_wmma_f32_16x16x32_bf16(false, va[ct][0].v, false, pop0.v,
                                                       (short)0, Ot[ct], false, false);
      Ot[ct] = __builtin_amdgcn_wmma_f32_16x16x32_bf16(false, va[ct][1].v, false, pop1.v,
                                                       (short)0, Ot[ct], false, false);
    }
    Ot4 = __builtin_amdgcn_wmma_f32_16x16x32_bf16(false, ones.v, false, pop0.v,
                                                  (short)0, Ot4, false, false);
    Ot4 = __builtin_amdgcn_wmma_f32_16x16x32_bf16(false, ones.v, false, pop1.v,
                                                  (short)0, Ot4, false, false);
  }

  float l_tot = Ot4[0] + xhalf_f32(Ot4[0]);   // half0 holds l, half1 holds 0
  float inv = 1.0f / l_tot;
  float g = gamma[0];
  size_t obase = (size_t)b * (size_t)kN * kC + (size_t)q * kC;
#pragma unroll
  for (int ct = 0; ct < 4; ++ct) {
    int cb = ct * 16 + 8 * half;                 // channels cb..cb+7
    const float4 x0 = *(const float4*)(x + obase + cb);
    const float4 x1 = *(const float4*)(x + obase + cb + 4);
    float4 o0, o1;
    o0.x = fmaf(g, Ot[ct][0] * inv, x0.x);
    o0.y = fmaf(g, Ot[ct][1] * inv, x0.y);
    o0.z = fmaf(g, Ot[ct][2] * inv, x0.z);
    o0.w = fmaf(g, Ot[ct][3] * inv, x0.w);
    o1.x = fmaf(g, Ot[ct][4] * inv, x1.x);
    o1.y = fmaf(g, Ot[ct][5] * inv, x1.y);
    o1.z = fmaf(g, Ot[ct][6] * inv, x1.z);
    o1.w = fmaf(g, Ot[ct][7] * inv, x1.w);
    *(float4*)(out + obase + cb)     = o0;
    *(float4*)(out + obase + cb + 4) = o1;
  }
}

extern "C" void kernel_launch(void* const* d_in, const int* in_sizes, int n_in,
                              void* d_out, int out_size, void* d_ws, size_t ws_size,
                              hipStream_t stream) {
  (void)in_sizes; (void)n_in; (void)out_size; (void)ws_size;
  const float* x     = (const float*)d_in[0];
  const float* Wq    = (const float*)d_in[1];
  const float* bq    = (const float*)d_in[2];
  const float* Wk    = (const float*)d_in[3];
  const float* bk    = (const float*)d_in[4];
  const float* Wv    = (const float*)d_in[5];
  const float* bv    = (const float*)d_in[6];
  const float* gamma = (const float*)d_in[7];
  float* out = (float*)d_out;

  // Workspace: q [B][N][8] | k [B][N][8] | v^T [B][C][N] | 16B zero pad (bf16)
  unsigned short* qws  = (unsigned short*)d_ws;
  unsigned short* kws  = qws + (size_t)kB * kN * kD;
  unsigned short* vtws = kws + (size_t)kB * kN * kD;
  unsigned short* zws  = vtws + (size_t)kB * kC * kN;

  proj_qkv_kernel<<<dim3(kN / 64, kB), 256, 0, stream>>>(
      x, Wq, bq, Wk, bk, Wv, bv, qws, kws, vtws, zws);
  flash_attn_kernel<<<dim3(kN / 64, kB), 128, 0, stream>>>(
      x, gamma, qws, kws, vtws, zws, out);
}
